// GraphQNetwork_22832046145746
// MI455X (gfx1250) — matrix operand order repaired and checked by reading, not compiled, not run on hardware
//
#include <hip/hip_runtime.h>
#include <hip/hip_fp16.h>
#include <math.h>

typedef __attribute__((ext_vector_type(16))) _Float16 v16h;
typedef __attribute__((ext_vector_type(8)))  float    v8f;

#define HEADS 4
#define CH    8
#define HID   32
#define ACTN  8
#define NGRAPH 256

// ---------------- utility fills (graph-capture safe, deterministic) ---------
__global__ void fill_f32(float* p, float v, int n) {
    int i = blockIdx.x * blockDim.x + threadIdx.x;
    if (i < n) p[i] = v;
}
__global__ void fill_u32(unsigned* p, unsigned v, int n) {
    int i = blockIdx.x * blockDim.x + threadIdx.x;
    if (i < n) p[i] = v;
}

// order-preserving float<->uint mapping for atomicMax over signed floats
__device__ __forceinline__ unsigned fenc(float f) {
    unsigned u = __float_as_uint(f);
    return (u & 0x80000000u) ? ~u : (u | 0x80000000u);
}
__device__ __forceinline__ float fdec(unsigned u) {
    return (u & 0x80000000u) ? __uint_as_float(u & 0x7fffffffu)
                             : __uint_as_float(~u);
}
__device__ __forceinline__ float lrelu02(float x) { return x > 0.f ? x : 0.2f * x; }

// ---------------- node projection: H = X[K] @ W[K,32] via WMMA -------------
// One wave computes a 16-node x 32-col tile; fp16 inputs, fp32 accumulate.
// B-fragments for W are pre-swizzled into LDS once per block: each lane then
// reads its whole 16-half fragment as a single 32B ds_load.
template <int K>
__global__ void proj_wmma(const float* __restrict__ X, const float* __restrict__ W,
                          float* __restrict__ Hout, int nTiles, int nNodes) {
    constexpr int NFRAG = (K / 32) * 2;             // k-steps * 2 column tiles
    __shared__ v16h sB[NFRAG * 32];                 // [frag][lane] fragments
    {
        _Float16* sBh = (_Float16*)sB;
        for (int i = threadIdx.x; i < NFRAG * 32 * 16; i += blockDim.x) {
            int j    = i & 15;          // half index within fragment
            int ln   = (i >> 4) & 31;   // lane
            int frag = i >> 9;          // g*2 + coltile
            int g    = frag >> 1;
            int ct   = frag & 1;
            int hi   = ln >> 4;
            int nc   = ln & 15;
            int k    = g * 32 + hi * 8 + (j & 7) + (j >> 3) * 16; // B layout: halves0-7 K 8h+j,
            sBh[i] = (_Float16)W[k * HID + ct * 16 + nc]; // halves8-15 K 16+8h+(j-8)
        }
    }
    __syncthreads();

    const int wave = threadIdx.x >> 5;
    const int lane = threadIdx.x & 31;
    const int t = blockIdx.x * (blockDim.x >> 5) + wave;
    if (t >= nTiles) return;              // wave-uniform: EXEC stays all-1s
    const int m0 = t * 16;
    const int mrow = m0 + (lane & 15);
    const int mr = mrow < nNodes ? mrow : (nNodes - 1);
    const int hi = lane >> 4;             // lane group 0/1
    const int nc = lane & 15;

    v8f c0 = {}; v8f c1 = {};
#pragma unroll
    for (int g = 0; g < K / 32; ++g) {
        const int k0 = g * 32;
        // A fragment: 16x32 f16. lanes0-15: halves0-7 => K 0-7, 8-15 => K 16-23
        //             lanes16-31: halves0-7 => K 8-15, 8-15 => K 24-31
        // Per lane that is two contiguous 8-float spans -> four float4 loads.
        const float* xr = X + (size_t)mr * K + k0 + hi * 8;
        float4 x0 = *(const float4*)(xr);
        float4 x1 = *(const float4*)(xr + 4);
        float4 x2 = *(const float4*)(xr + 16);
        float4 x3 = *(const float4*)(xr + 20);
        v16h a;
        a[0]=(_Float16)x0.x; a[1]=(_Float16)x0.y; a[2]=(_Float16)x0.z; a[3]=(_Float16)x0.w;
        a[4]=(_Float16)x1.x; a[5]=(_Float16)x1.y; a[6]=(_Float16)x1.z; a[7]=(_Float16)x1.w;
        a[8]=(_Float16)x2.x; a[9]=(_Float16)x2.y; a[10]=(_Float16)x2.z; a[11]=(_Float16)x2.w;
        a[12]=(_Float16)x3.x; a[13]=(_Float16)x3.y; a[14]=(_Float16)x3.z; a[15]=(_Float16)x3.w;

        v16h b0 = sB[(g * 2 + 0) * 32 + lane];
        v16h b1 = sB[(g * 2 + 1) * 32 + lane];
        c0 = __builtin_amdgcn_wmma_f32_16x16x32_f16(false, a, false, b0,
                                                    (short)0, c0, false, false);
        c1 = __builtin_amdgcn_wmma_f32_16x16x32_f16(false, a, false, b1,
                                                    (short)0, c1, false, false);
    }
    // D layout: VGPR r -> rows (r, r+8) for lane groups (0,1); col = lane&15
#pragma unroll
    for (int r = 0; r < 8; ++r) {
        int row = m0 + r + hi * 8;
        if (row < nNodes) {
            Hout[row * HID + nc]      = c0[r];
            Hout[row * HID + 16 + nc] = c1[r];
        }
    }
}

// ---------------- per-node attention logits --------------------------------
__global__ void att_logits(const float* __restrict__ Hn,
                           const float* __restrict__ att_s,
                           const float* __restrict__ att_d,
                           float* __restrict__ als, float* __restrict__ ald, int n) {
    int i = blockIdx.x * blockDim.x + threadIdx.x;
    if (i >= n) return;
    const float4* hv = (const float4*)(Hn + (size_t)i * HID);
#pragma unroll
    for (int hd = 0; hd < HEADS; ++hd) {
        float4 a = hv[hd * 2], b = hv[hd * 2 + 1];
        const float4* s4 = (const float4*)(att_s + hd * CH);
        const float4* d4 = (const float4*)(att_d + hd * CH);
        float4 sa = s4[0], sb = s4[1], da = d4[0], db = d4[1];
        float vs = a.x*sa.x + a.y*sa.y + a.z*sa.z + a.w*sa.w +
                   b.x*sb.x + b.y*sb.y + b.z*sb.z + b.w*sb.w;
        float vd = a.x*da.x + a.y*da.y + a.z*da.z + a.w*da.w +
                   b.x*db.x + b.y*db.y + b.z*db.z + b.w*db.w;
        als[i * HEADS + hd] = vs;
        ald[i * HEADS + hd] = vd;
    }
}

// ---------------- edge helpers ---------------------------------------------
__device__ __forceinline__ void edge_sd(const int* __restrict__ ei, int E, int i,
                                        int& s, int& d) {
    if (i < E) { s = ei[i]; d = ei[E + i]; }
    else       { s = d = i - E; }          // self-loops appended
}
__device__ __forceinline__ void edge_e(const float* __restrict__ als,
                                       const float* __restrict__ ald,
                                       int s, int d, float e[HEADS]) {
    float4 as = *(const float4*)(als + (size_t)s * HEADS);
    float4 ad = *(const float4*)(ald + (size_t)d * HEADS);
    e[0] = lrelu02(as.x + ad.x);
    e[1] = lrelu02(as.y + ad.y);
    e[2] = lrelu02(as.z + ad.z);
    e[3] = lrelu02(as.w + ad.w);
}

// pass 1: segment max (encoded-uint atomicMax, L2-resident)
__global__ void edge_max(const int* __restrict__ ei, const float* __restrict__ als,
                         const float* __restrict__ ald, unsigned* __restrict__ m,
                         int E, int n) {
    int i = blockIdx.x * blockDim.x + threadIdx.x;
    if (i >= E + n) return;
    int s, d; edge_sd(ei, E, i, s, d);
    float e[HEADS]; edge_e(als, ald, s, d, e);
#pragma unroll
    for (int hd = 0; hd < HEADS; ++hd)
        atomicMax(&m[(size_t)d * HEADS + hd], fenc(e[hd]));
}

// pass 2: segment sum of exp(e - m)
__global__ void edge_sum(const int* __restrict__ ei, const float* __restrict__ als,
                         const float* __restrict__ ald, const unsigned* __restrict__ m,
                         float* __restrict__ sArr, int E, int n) {
    int i = blockIdx.x * blockDim.x + threadIdx.x;
    if (i >= E + n) return;
    int s, d; edge_sd(ei, E, i, s, d);
    float e[HEADS]; edge_e(als, ald, s, d, e);
#pragma unroll
    for (int hd = 0; hd < HEADS; ++hd) {
        float p = __expf(e[hd] - fdec(m[(size_t)d * HEADS + hd]));
        atomicAdd(&sArr[(size_t)d * HEADS + hd], p);
    }
}

// pass 3: weighted scatter of h[src] into agg[dst].
// One thread per (edge, head): 4x more threads than per-edge, each doing two
// float4 gathers + 8 atomics -> better latency hiding on the L2 atomic units.
__global__ void edge_acc(const int* __restrict__ ei, const float* __restrict__ als,
                         const float* __restrict__ ald, const unsigned* __restrict__ m,
                         const float* __restrict__ sArr, const float* __restrict__ Hn,
                         float* __restrict__ agg, int E, int n) {
    int i = blockIdx.x * blockDim.x + threadIdx.x;
    if (i >= (E + n) * HEADS) return;
    int eidx = i >> 2;
    int hd   = i & (HEADS - 1);
    int s, d; edge_sd(ei, E, eidx, s, d);
    float e = lrelu02(als[(size_t)s * HEADS + hd] + ald[(size_t)d * HEADS + hd]);
    float p = __expf(e - fdec(m[(size_t)d * HEADS + hd]));
    float a = p / sArr[(size_t)d * HEADS + hd];
    const float* hs = Hn + (size_t)s * HID + hd * CH;
    float* ag = agg + (size_t)d * HID + hd * CH;
    float4 h0 = *(const float4*)(hs);
    float4 h1 = *(const float4*)(hs + 4);
    atomicAdd(&ag[0], h0.x * a);
    atomicAdd(&ag[1], h0.y * a);
    atomicAdd(&ag[2], h0.z * a);
    atomicAdd(&ag[3], h0.w * a);
    atomicAdd(&ag[4], h1.x * a);
    atomicAdd(&ag[5], h1.y * a);
    atomicAdd(&ag[6], h1.z * a);
    atomicAdd(&ag[7], h1.w * a);
}

// bias + ELU
__global__ void bias_elu(const float* __restrict__ agg, const float* __restrict__ b,
                         float* __restrict__ act, int total) {
    int i = blockIdx.x * blockDim.x + threadIdx.x;
    if (i >= total) return;
    float v = agg[i] + b[i & (HID - 1)];
    act[i] = v > 0.f ? v : (__expf(v) - 1.f);
}

// global mean pool (sum + count)
__global__ void pool(const float* __restrict__ act, const int* __restrict__ batch,
                     float* __restrict__ gsum, float* __restrict__ cnt, int n) {
    int i = blockIdx.x * blockDim.x + threadIdx.x;
    if (i >= n * HID) return;
    int node = i >> 5, c = i & (HID - 1);
    int g = batch[node];
    atomicAdd(&gsum[(size_t)g * HID + c], act[i]);
    if (c == 0) atomicAdd(&cnt[g], 1.0f);
}

// dueling head: one 32-thread block per graph
__global__ void head(const float* __restrict__ gsum, const float* __restrict__ cnt,
                     const float* __restrict__ vsW, const float* __restrict__ vsb,
                     const float* __restrict__ vW, const float* __restrict__ vb,
                     const float* __restrict__ advW, const float* __restrict__ advb,
                     const float* __restrict__ aW, const float* __restrict__ ab,
                     float* __restrict__ out) {
    __shared__ float g[HID], hv[HID], ha[HID], advs[ACTN], val[1];
    int j = threadIdx.x, gi = blockIdx.x;
    float c = cnt[gi]; c = c > 1.f ? c : 1.f;
    g[j] = gsum[(size_t)gi * HID + j] / c;
    __syncthreads();
    float sv = vsb[j], sa = advb[j];
    for (int k = 0; k < HID; ++k) {
        float gk = g[k];
        sv += gk * vsW[k * HID + j];
        sa += gk * advW[k * HID + j];
    }
    hv[j] = sv > 0.f ? sv : 0.f;
    ha[j] = sa > 0.f ? sa : 0.f;
    __syncthreads();
    if (j < ACTN) {
        float s = ab[j];
        for (int k = 0; k < HID; ++k) s += ha[k] * aW[k * ACTN + j];
        advs[j] = s;
    }
    if (j == 0) {
        float v = vb[0];
        for (int k = 0; k < HID; ++k) v += hv[k] * vW[k];
        val[0] = v;
    }
    __syncthreads();
    if (j == 0) {
        float mean = 0.f;
        for (int k = 0; k < ACTN; ++k) mean += advs[k];
        mean *= (1.0f / ACTN);
        for (int k = 0; k < ACTN; ++k)
            out[(size_t)gi * ACTN + k] = val[0] + advs[k] - mean;
    }
}

// ---------------------------------------------------------------------------
extern "C" void kernel_launch(void* const* d_in, const int* in_sizes, int n_in,
                              void* d_out, int out_size, void* d_ws, size_t ws_size,
                              hipStream_t stream) {
    const float* x        = (const float*)d_in[0];
    const int*   ei       = (const int*)d_in[1];
    const int*   batch    = (const int*)d_in[2];
    const float* W1       = (const float*)d_in[3];
    const float* att_s1   = (const float*)d_in[4];
    const float* att_d1   = (const float*)d_in[5];
    const float* b1       = (const float*)d_in[6];
    const float* W2       = (const float*)d_in[7];
    const float* att_s2   = (const float*)d_in[8];
    const float* att_d2   = (const float*)d_in[9];
    const float* b2       = (const float*)d_in[10];
    const float* vsW      = (const float*)d_in[11];
    const float* vsb      = (const float*)d_in[12];
    const float* vW       = (const float*)d_in[13];
    const float* vb       = (const float*)d_in[14];
    const float* advW     = (const float*)d_in[15];
    const float* advb     = (const float*)d_in[16];
    const float* aW       = (const float*)d_in[17];
    const float* ab       = (const float*)d_in[18];
    float* out = (float*)d_out;

    const int N = in_sizes[0] / 64;   // F_IN = 64
    const int E = in_sizes[1] / 2;
    const int Etot = E + N;

    // workspace layout (all 4-byte elements)
    float* f = (float*)d_ws;
    size_t o = 0;
    float*    buf_h   = f + o; o += (size_t)N * HID;
    float*    buf_agg = f + o; o += (size_t)N * HID;
    float*    buf_act = f + o; o += (size_t)N * HID;
    float*    buf_als = f + o; o += (size_t)N * HEADS;
    float*    buf_ald = f + o; o += (size_t)N * HEADS;
    unsigned* buf_m   = (unsigned*)(f + o); o += (size_t)N * HEADS;
    float*    buf_s   = f + o; o += (size_t)N * HEADS;
    float*    gsum    = f + o; o += (size_t)NGRAPH * HID;
    float*    cnt     = f + o; o += NGRAPH;
    (void)ws_size;

    const int BLK = 256;
    const int nTiles = (N + 15) / 16;
    const int projBlocks = (nTiles + 7) / 8;          // 8 waves per block
    const int nodeBlocks = (N + BLK - 1) / BLK;
    const int edgeBlocks = (Etot + BLK - 1) / BLK;
    const int accBlocks  = ((size_t)Etot * HEADS + BLK - 1) / BLK;
    const int chanBlocks = ((size_t)N * HID + BLK - 1) / BLK;
    const int nhBlocks   = ((size_t)N * HEADS + BLK - 1) / BLK;

    // ---------------- layer 1 ----------------
    fill_u32<<<nhBlocks, BLK, 0, stream>>>(buf_m, 0u, N * HEADS);
    fill_f32<<<nhBlocks, BLK, 0, stream>>>(buf_s, 0.f, N * HEADS);
    fill_f32<<<chanBlocks, BLK, 0, stream>>>(buf_agg, 0.f, N * HID);

    proj_wmma<64><<<projBlocks, BLK, 0, stream>>>(x, W1, buf_h, nTiles, N);
    att_logits<<<nodeBlocks, BLK, 0, stream>>>(buf_h, att_s1, att_d1,
                                               buf_als, buf_ald, N);
    edge_max<<<edgeBlocks, BLK, 0, stream>>>(ei, buf_als, buf_ald, buf_m, E, N);
    edge_sum<<<edgeBlocks, BLK, 0, stream>>>(ei, buf_als, buf_ald, buf_m, buf_s, E, N);
    edge_acc<<<accBlocks, BLK, 0, stream>>>(ei, buf_als, buf_ald, buf_m, buf_s,
                                            buf_h, buf_agg, E, N);
    bias_elu<<<chanBlocks, BLK, 0, stream>>>(buf_agg, b1, buf_act, N * HID);

    // ---------------- layer 2 ----------------
    fill_u32<<<nhBlocks, BLK, 0, stream>>>(buf_m, 0u, N * HEADS);
    fill_f32<<<nhBlocks, BLK, 0, stream>>>(buf_s, 0.f, N * HEADS);
    fill_f32<<<chanBlocks, BLK, 0, stream>>>(buf_agg, 0.f, N * HID);

    proj_wmma<32><<<projBlocks, BLK, 0, stream>>>(buf_act, W2, buf_h, nTiles, N);
    att_logits<<<nodeBlocks, BLK, 0, stream>>>(buf_h, att_s2, att_d2,
                                               buf_als, buf_ald, N);
    edge_max<<<edgeBlocks, BLK, 0, stream>>>(ei, buf_als, buf_ald, buf_m, E, N);
    edge_sum<<<edgeBlocks, BLK, 0, stream>>>(ei, buf_als, buf_ald, buf_m, buf_s, E, N);
    edge_acc<<<accBlocks, BLK, 0, stream>>>(ei, buf_als, buf_ald, buf_m, buf_s,
                                            buf_h, buf_agg, E, N);
    bias_elu<<<chanBlocks, BLK, 0, stream>>>(buf_agg, b2, buf_act, N * HID);

    // ---------------- pool + head ----------------
    fill_f32<<<(NGRAPH * HID + BLK - 1) / BLK, BLK, 0, stream>>>(gsum, 0.f, NGRAPH * HID);
    fill_f32<<<1, BLK, 0, stream>>>(cnt, 0.f, NGRAPH);
    pool<<<chanBlocks, BLK, 0, stream>>>(buf_act, batch, gsum, cnt, N);
    head<<<NGRAPH, HID, 0, stream>>>(gsum, cnt, vsW, vsb, vW, vb,
                                     advW, advb, aW, ab, out);
}
